// Linear_27608049779368
// MI455X (gfx1250) — matrix operand order repaired and checked by reading, not compiled, not run on hardware
//
#include <hip/hip_runtime.h>

typedef __attribute__((ext_vector_type(16))) _Float16 v16h;
typedef __attribute__((ext_vector_type(8)))  float    v8f;
typedef __attribute__((ext_vector_type(2)))  unsigned int v2u;

#define BATCH 4096
#define FEAT  32
#define NCLS  128
#define N1    32
#define N2    496
#define N3    4960
#define NTOT  5488            // 32 + 496 + 4960
#define KCHUNKS 172           // ceil(5488 / 32)
#define KPAD  (KCHUNKS * 32)  // 5504

// workspace layout (bytes)
#define WS_IDX_OFF 0          // KPAD * 4 = 22016 B of packed combo ids
#define WS_W_OFF   22528      // 256-aligned; KCHUNKS*8*32*16 f16 = 1,409,024 B

// ---------------------------------------------------------------------------
// Prolog 1: pack each combo (order 1..3) into one u32 of three 6-bit ids.
// Feature id 32 is a synthetic "ones" column -> identity multiplier.
// ---------------------------------------------------------------------------
__global__ void build_packed_idx(const int* __restrict__ idx1,
                                 const int* __restrict__ idx2,
                                 const int* __restrict__ idx3,
                                 unsigned int* __restrict__ pidx) {
    int t = blockIdx.x * blockDim.x + threadIdx.x;
    if (t >= KPAD) return;
    unsigned int f0 = 32u, f1 = 32u, f2 = 32u;
    if (t < N1) {
        f0 = (unsigned int)idx1[t];
    } else if (t < N1 + N2) {
        int j = t - N1;
        f0 = (unsigned int)idx2[2 * j];
        f1 = (unsigned int)idx2[2 * j + 1];
    } else if (t < NTOT) {
        int j = t - (N1 + N2);
        f0 = (unsigned int)idx3[3 * j];
        f1 = (unsigned int)idx3[3 * j + 1];
        f2 = (unsigned int)idx3[3 * j + 2];
    } // t >= NTOT: all-ones product, but W rows are zero-padded -> contributes 0
    pidx[t] = f0 | (f1 << 6) | (f2 << 12);
}

// ---------------------------------------------------------------------------
// Prolog 2: convert stacked W (f32) to f16 in this code's contiguous-K WMMA B-operand layout:
//   B is 32x16 (KxN); lanes 0-15 hold col n=lane with halfs = K 0..15,
//   lanes 16-31 hold col n=lane-16 with halfs = K 16..31.
// Linear index: e = ((kc*8 + ct)*32 + lane)*16 + half
// ---------------------------------------------------------------------------
__global__ void swizzle_w(const float* __restrict__ W1,
                          const float* __restrict__ W2,
                          const float* __restrict__ W3,
                          _Float16* __restrict__ wswz) {
    int e = blockIdx.x * blockDim.x + threadIdx.x;
    if (e >= KCHUNKS * 8 * 32 * 16) return;
    int h    = e & 15;
    int lane = (e >> 4) & 31;
    int ct   = (e >> 9) & 7;
    int kc   = e >> 12;
    int k    = kc * 32 + ((lane >> 4) * 16) + h;
    int col  = ct * 16 + (lane & 15);
    float v = 0.0f;
    if (k < N1)            v = W1[k * NCLS + col];
    else if (k < N1 + N2)  v = W2[(k - N1) * NCLS + col];
    else if (k < NTOT)     v = W3[(k - (N1 + N2)) * NCLS + col];
    wswz[e] = (_Float16)v;
}

// ---------------------------------------------------------------------------
// Main GEMM: out[4096,128] = P[4096,5488] @ W[5488,128] + bias
// P built on the fly per 16x32 A tile from an LDS-transposed x tile.
// 64 WGs x 256 threads (8 wave32); each wave: 16 rows x 64 cols (4 WMMA tiles).
// ---------------------------------------------------------------------------
__global__ __launch_bounds__(256) void fm_wmma(const float* __restrict__ x,
                                               const float* __restrict__ bias,
                                               const unsigned int* __restrict__ pidx,
                                               const _Float16* __restrict__ wswz,
                                               float* __restrict__ out) {
    // x transposed tile: sXT[f*65 + r], stride 65 -> conflict-free on both
    // the transpose store (consecutive f) and the gather read (consecutive r).
    __shared__ float        sXT[33 * 65];   // 33 "features": 32 real + ones col
    __shared__ unsigned int sIdx[KPAD];

    const int tid     = threadIdx.x;
    const int rowbase = blockIdx.x * 64;

    // coalesced load of x[rowbase .. rowbase+63][0..31], transposed into LDS
    for (int i = tid; i < 64 * FEAT; i += 256) {
        int r = i >> 5;   // 0..63
        int f = i & 31;   // 0..31
        sXT[f * 65 + r] = x[(rowbase + r) * FEAT + f];
    }
    if (tid < 64) sXT[32 * 65 + tid] = 1.0f;   // identity column (id 32)
    for (int i = tid; i < KPAD; i += 256) sIdx[i] = pidx[i];
    __syncthreads();

    const int lane   = tid & 31;
    const int wave   = tid >> 5;
    const int stripe = wave >> 1;      // 0..3  : 16-row stripe in this block
    const int halfc  = wave & 1;       // 0/1   : which 64 output columns
    const int g      = lane >> 4;      // lane group (K-half select for A)
    const int m16    = lane & 15;
    // per-lane row base pointer: gather address is xr[f * 65]
    const float* xr  = sXT + (stripe * 16 + m16);

    v8f acc[4] = {};   // 4 column tiles of 16x16 f32

    for (int kc = 0; kc < KCHUNKS; ++kc) {
        // ---- build A operand (16x32 f16) in contiguous-K order, matching B ----
        // lanes 0-15:  VGPR v -> K = 2v,2v+1
        // lanes 16-31: same pattern shifted by +16
        v16h a;
        const int c0 = kc * 32 + 16 * g;
#pragma unroll
        for (int v = 0; v < 8; ++v) {
            const int base = c0 + 2 * v;   // always even
            const v2u p = *(const v2u*)&sIdx[base];             // ds_load_b64
            const unsigned int p0 = p.x, p1 = p.y;
            const float q0 = xr[(p0 & 63u) * 65]
                           * xr[((p0 >> 6) & 63u) * 65]
                           * xr[((p0 >> 12) & 63u) * 65];
            const float q1 = xr[(p1 & 63u) * 65]
                           * xr[((p1 >> 6) & 63u) * 65]
                           * xr[((p1 >> 12) & 63u) * 65];
            a[2 * v]     = (_Float16)q0;
            a[2 * v + 1] = (_Float16)q1;
        }

        // ---- pre-swizzled B tiles: one 32B vector load per lane per tile ----
        const _Float16* wb = wswz + ((size_t)(kc * 8 + halfc * 4) * 32 + lane) * 16;
        const v16h b0 = *(const v16h*)(wb);
        const v16h b1 = *(const v16h*)(wb + 1 * 32 * 16);
        const v16h b2 = *(const v16h*)(wb + 2 * 32 * 16);
        const v16h b3 = *(const v16h*)(wb + 3 * 32 * 16);
        // prefetch the B stream two chunks (16 KB) ahead -> global_prefetch_b8
        __builtin_prefetch(wb + 2 * 8 * 32 * 16, 0, 1);

        acc[0] = __builtin_amdgcn_wmma_f32_16x16x32_f16(false, a, false, b0,
                                                        (short)0, acc[0], false, false);
        acc[1] = __builtin_amdgcn_wmma_f32_16x16x32_f16(false, a, false, b1,
                                                        (short)0, acc[1], false, false);
        acc[2] = __builtin_amdgcn_wmma_f32_16x16x32_f16(false, a, false, b2,
                                                        (short)0, acc[2], false, false);
        acc[3] = __builtin_amdgcn_wmma_f32_16x16x32_f16(false, a, false, b3,
                                                        (short)0, acc[3], false, false);
    }

    // ---- epilogue: D layout is lanes0-15 M=j / lanes16-31 M=8+j, col=lane%16
#pragma unroll
    for (int t = 0; t < 4; ++t) {
        const int col = (halfc * 4 + t) * 16 + m16;
        const float bv = bias[col];
#pragma unroll
        for (int j = 0; j < 8; ++j) {
            const int row = rowbase + stripe * 16 + 8 * g + j;
            out[row * NCLS + col] = acc[t][j] + bv;
        }
    }
}

extern "C" void kernel_launch(void* const* d_in, const int* in_sizes, int n_in,
                              void* d_out, int out_size, void* d_ws, size_t ws_size,
                              hipStream_t stream) {
    const float* x    = (const float*)d_in[0];
    const float* bias = (const float*)d_in[1];
    const float* W1   = (const float*)d_in[2];
    const float* W2   = (const float*)d_in[3];
    const float* W3   = (const float*)d_in[4];
    const int*   idx1 = (const int*)d_in[5];
    const int*   idx2 = (const int*)d_in[6];
    const int*   idx3 = (const int*)d_in[7];

    unsigned int* pidx = (unsigned int*)((char*)d_ws + WS_IDX_OFF);
    _Float16*     wswz = (_Float16*)((char*)d_ws + WS_W_OFF);

    build_packed_idx<<<(KPAD + 255) / 256, 256, 0, stream>>>(idx1, idx2, idx3, pidx);

    const int wtot = KCHUNKS * 8 * 32 * 16;
    swizzle_w<<<(wtot + 255) / 256, 256, 0, stream>>>(W1, W2, W3, wswz);

    fm_wmma<<<BATCH / 64, 256, 0, stream>>>(x, bias, pidx, wswz, (float*)d_out);
}
